// TrajectoryInformedSplatAttention_43224550867598
// MI455X (gfx1250) — matrix operand order repaired and checked by reading, not compiled, not run on hardware
//
#include <hip/hip_runtime.h>
#include <cstdint>
#include <cstddef>

// ---------------------------------------------------------------------------
// TrajectoryInformedSplatAttention for MI455X (gfx1250, wave32, WMMA)
// B=4, S=2048, D=1024, G=512, W=8.
// Key reduction: windows are overlapping diff rows; all heavy math depends
// only on (b, t). Two bf16 WMMA GEMMs over 8188 unique rows + cheap 8-tap
// temporal blend.
// ---------------------------------------------------------------------------

#define B_  4
#define S_  2048
#define D_  1024
#define G_  512
#define T_  (S_ - 1)   // 2047 diff rows per batch
#define W_  8

typedef float  v8f   __attribute__((ext_vector_type(8)));
typedef __bf16 v16bf __attribute__((ext_vector_type(16)));

static __device__ __forceinline__ unsigned short f2bf(float f) {
  union { float f; uint32_t u; } a; a.f = f;
  uint32_t r = a.u + 0x7FFFu + ((a.u >> 16) & 1u);   // round-to-nearest-even
  return (unsigned short)(r >> 16);
}
static __device__ __forceinline__ float bf2f(unsigned short s) {
  union { uint32_t u; float f; } a; a.u = ((uint32_t)s) << 16;
  return a.f;
}

// ---------------------------------------------------------------------------
// Kernel 0: guidance preprocessing.
//  gscale[g] = 2 / max(||guidance[g]||, 1e-8)   (softmax scale folded in)
//  gbf [g][d] = bf16(guidance[g][d])            (B operand of GEMM1)
//  gbfT[d][g] = bf16(guidance[g][d])            (B operand of GEMM2)
// ---------------------------------------------------------------------------
__global__ __launch_bounds__(256) void tisa_prep(
    const float* __restrict__ guid, float* __restrict__ gscale,
    unsigned short* __restrict__ gbf, unsigned short* __restrict__ gbfT)
{
  __shared__ float red[256];
  const int g = blockIdx.x;
  const int tid = threadIdx.x;
  const float* row = guid + (size_t)g * D_;
  float ssq = 0.0f;
  for (int i = tid; i < D_; i += 256) {
    float v = row[i];
    ssq += v * v;
    unsigned short hb = f2bf(v);
    gbf[(size_t)g * D_ + i] = hb;
    gbfT[(size_t)i * G_ + g] = hb;
  }
  red[tid] = ssq;
  __syncthreads();
  for (int o = 128; o > 0; o >>= 1) {
    if (tid < o) red[tid] += red[tid + o];
    __syncthreads();
  }
  if (tid == 0) gscale[g] = 2.0f / fmaxf(sqrtf(red[0]), 1e-8f);
}

// ---------------------------------------------------------------------------
// Kernel 1: per (b, t) rows — normalize diffs, cosine-sim softmax vs guidance
// (WMMA GEMM1), influence (WMMA GEMM2), blended row out.
// One workgroup = 8 wave32 handles a 16-row t-tile for one batch.
// LDS: 32KB bf16 normed tile + 32KB sims buffer (reused in-place for probs).
// ---------------------------------------------------------------------------
__global__ __launch_bounds__(256) void tisa_main(
    const float* __restrict__ emb,
    const unsigned short* __restrict__ gbf,
    const unsigned short* __restrict__ gbfT,
    const float* __restrict__ gscale,
    float* __restrict__ blended,
    float* __restrict__ mwOut)
{
  __shared__ unsigned short normedH[16 * D_];  // 32 KB bf16 normalized diffs
  __shared__ float simsL[16 * G_];             // 32 KB logits; reused as bf16 probs

  const int tid  = threadIdx.x;
  const int lane = tid & 31;
  const int wv   = tid >> 5;      // wave 0..7
  const int h    = lane >> 4;     // lane half
  const int l16  = lane & 15;
  const int b    = blockIdx.y;
  const int t0   = blockIdx.x * 16;

  // ---- Stage A: diffs -> L2-normalized rows (bf16) + magnitude weight ----
  {
    const int m  = 2 * wv + h;    // tile row 0..15 (one 16-lane group per row)
    const int t  = t0 + m;
    const int c0 = l16 * 64;      // 64 contiguous cols per lane
    const float* e0 = emb + ((size_t)b * S_ + t) * D_ + c0;
    float ssq = 0.0f;
    if (t < T_) {
      #pragma unroll
      for (int i = 0; i < 64; i += 4) {
        float4 a = *(const float4*)(e0 + i);
        float4 c = *(const float4*)(e0 + D_ + i);
        float dx = c.x - a.x, dy = c.y - a.y, dz = c.z - a.z, dw = c.w - a.w;
        ssq += dx * dx + dy * dy + dz * dz + dw * dw;
      }
    }
    #pragma unroll
    for (int off = 1; off < 16; off <<= 1) ssq += __shfl_xor(ssq, off, 32);
    const float mag = sqrtf(ssq);
    const float inv = (mag > 1e-6f) ? (1.0f / mag) : 0.0f;  // normed=0 if tiny
    if (l16 == 0 && t < T_) mwOut[b * T_ + t] = tanhf(2.0f * mag);
    if (t < T_) {
      #pragma unroll
      for (int i = 0; i < 64; i += 4) {
        float4 a = *(const float4*)(e0 + i);
        float4 c = *(const float4*)(e0 + D_ + i);
        normedH[m * D_ + c0 + i + 0] = f2bf((c.x - a.x) * inv);
        normedH[m * D_ + c0 + i + 1] = f2bf((c.y - a.y) * inv);
        normedH[m * D_ + c0 + i + 2] = f2bf((c.z - a.z) * inv);
        normedH[m * D_ + c0 + i + 3] = f2bf((c.w - a.w) * inv);
      }
    } else {
      #pragma unroll
      for (int i = 0; i < 64; ++i) normedH[m * D_ + c0 + i] = 0;
    }
  }
  __syncthreads();

  // ---- GEMM1: sims[m][g] = 2 * <normed[m], guidance[g]> / ||guidance[g]||
  // Each wave owns 64 g-columns (4 N-tiles); K = 1024 in 32 steps of 32.
  {
    const int gbase = wv * 64;
    v8f acc[4];
    #pragma unroll
    for (int nt = 0; nt < 4; ++nt) acc[nt] = 0.0f;

    for (int ks = 0; ks < 32; ++ks) {
      union { v16bf v; uint32_t u[8]; } A;
      #pragma unroll
      for (int j = 0; j < 8; ++j) {
        const int k = ks * 32 + ((j >= 4) ? 16 : 0) + h * 8 + (j & 3) * 2;
        A.u[j] = *(const uint32_t*)&normedH[l16 * D_ + k];
      }
      #pragma unroll
      for (int nt = 0; nt < 4; ++nt) {
        const int g = gbase + nt * 16 + l16;
        union { v16bf v; uint32_t u[8]; } Bf;
        #pragma unroll
        for (int j = 0; j < 8; ++j) {
          const int k = ks * 32 + ((j >= 4) ? 16 : 0) + h * 8 + (j & 3) * 2;
          Bf.u[j] = *(const uint32_t*)(gbf + (size_t)g * D_ + k);
        }
        acc[nt] = __builtin_amdgcn_wmma_f32_16x16x32_bf16(
            false, A.v, false, Bf.v, (short)0, acc[nt], false, false);
      }
    }
    #pragma unroll
    for (int nt = 0; nt < 4; ++nt) {
      const int g = gbase + nt * 16 + l16;
      const float gs = gscale[g];
      #pragma unroll
      for (int r = 0; r < 8; ++r) {
        const int m = r + 8 * h;           // C/D layout: M = r + 8*half
        simsL[m * G_ + g] = acc[nt][r] * gs;
      }
    }
  }
  __syncthreads();

  // ---- Softmax over G=512 per row (wave handles rows 2w, 2w+1) ----
  {
    const int m0 = 2 * wv, m1 = m0 + 1;
    float v0[16], v1[16];
    float mx0 = -3.4e38f, mx1 = -3.4e38f;
    #pragma unroll
    for (int i = 0; i < 16; ++i) {
      v0[i] = simsL[m0 * G_ + lane + 32 * i];
      v1[i] = simsL[m1 * G_ + lane + 32 * i];
      mx0 = fmaxf(mx0, v0[i]);
      mx1 = fmaxf(mx1, v1[i]);
    }
    #pragma unroll
    for (int off = 16; off >= 1; off >>= 1) {
      mx0 = fmaxf(mx0, __shfl_xor(mx0, off, 32));
      mx1 = fmaxf(mx1, __shfl_xor(mx1, off, 32));
    }
    float s0 = 0.0f, s1 = 0.0f;
    #pragma unroll
    for (int i = 0; i < 16; ++i) {
      v0[i] = __expf(v0[i] - mx0); s0 += v0[i];
      v1[i] = __expf(v1[i] - mx1); s1 += v1[i];
    }
    #pragma unroll
    for (int off = 16; off >= 1; off >>= 1) {
      s0 += __shfl_xor(s0, off, 32);
      s1 += __shfl_xor(s1, off, 32);
    }
    const float r0 = 1.0f / s0, r1 = 1.0f / s1;
    __syncthreads();                       // all sims reads done
    unsigned short* pH = (unsigned short*)simsL;
    #pragma unroll
    for (int i = 0; i < 16; ++i) {
      pH[m0 * G_ + lane + 32 * i] = f2bf(v0[i] * r0);
      pH[m1 * G_ + lane + 32 * i] = f2bf(v1[i] * r1);
    }
  }
  __syncthreads();

  // ---- GEMM2: influence[m][d] = sum_g p[m][g] * guidance[g][d];
  // blended = 0.6*normed + 0.4*influence. Each wave owns 128 d-columns.
  {
    const unsigned short* pH = (const unsigned short*)simsL;
    const int dbase = wv * 128;
    v8f acc[8];
    #pragma unroll
    for (int nt = 0; nt < 8; ++nt) acc[nt] = 0.0f;

    for (int ks = 0; ks < 16; ++ks) {      // K = 512 over g
      union { v16bf v; uint32_t u[8]; } A;
      #pragma unroll
      for (int j = 0; j < 8; ++j) {
        const int k = ks * 32 + ((j >= 4) ? 16 : 0) + h * 8 + (j & 3) * 2;
        A.u[j] = *(const uint32_t*)&pH[l16 * G_ + k];
      }
      #pragma unroll
      for (int nt = 0; nt < 8; ++nt) {
        const int d = dbase + nt * 16 + l16;
        union { v16bf v; uint32_t u[8]; } Bf;
        #pragma unroll
        for (int j = 0; j < 8; ++j) {
          const int k = ks * 32 + ((j >= 4) ? 16 : 0) + h * 8 + (j & 3) * 2;
          Bf.u[j] = *(const uint32_t*)(gbfT + (size_t)d * G_ + k);
        }
        acc[nt] = __builtin_amdgcn_wmma_f32_16x16x32_bf16(
            false, A.v, false, Bf.v, (short)0, acc[nt], false, false);
      }
    }
    #pragma unroll
    for (int nt = 0; nt < 8; ++nt) {
      const int d = dbase + nt * 16 + l16;
      #pragma unroll
      for (int r = 0; r < 8; ++r) {
        const int m = r + 8 * h;
        const int t = t0 + m;
        if (t < T_) {
          const float nv = bf2f(normedH[m * D_ + d]);
          blended[((size_t)(b * T_ + t)) * D_ + d] = 0.6f * nv + 0.4f * acc[nt][r];
        }
      }
    }
  }
}

// ---------------------------------------------------------------------------
// Kernel 2: 8-tap temporal blend.
// out[b,s,:] = sum_k posw(s,k)*mw[b,s-8+k]*blended[b,s-8+k,:] / max(sumw,1e-8)
// ---------------------------------------------------------------------------
__global__ __launch_bounds__(256) void tisa_finalize(
    const float* __restrict__ blended, const float* __restrict__ mw,
    float* __restrict__ out)
{
  const int s = blockIdx.x;
  const int b = blockIdx.y;
  const int d = threadIdx.x * 4;
  float4 acc = make_float4(0.f, 0.f, 0.f, 0.f);
  float wsum = 0.0f;
  if (s >= 1) {
    const int w = (s < W_) ? s : W_;
    #pragma unroll
    for (int k = 0; k < W_; ++k) {
      if (k < W_ - w) continue;            // invalid (idx < 0)
      const int idx = s - W_ + k;
      const int j = k - (W_ - w);
      const float lin = (w > 1) ? (0.1f + 0.9f * (float)j / (float)(w - 1)) : 0.1f;
      const float wt = lin * mw[b * T_ + idx];
      const float4 v = *(const float4*)(blended + ((size_t)(b * T_ + idx)) * D_ + d);
      acc.x += wt * v.x; acc.y += wt * v.y; acc.z += wt * v.z; acc.w += wt * v.w;
      wsum += wt;
    }
  }
  const float r = 1.0f / fmaxf(wsum, 1e-8f);
  float4 o = make_float4(acc.x * r, acc.y * r, acc.z * r, acc.w * r);
  *(float4*)(out + ((size_t)(b * S_ + s)) * D_ + d) = o;
}

// ---------------------------------------------------------------------------
extern "C" void kernel_launch(void* const* d_in, const int* in_sizes, int n_in,
                              void* d_out, int out_size, void* d_ws, size_t ws_size,
                              hipStream_t stream)
{
  (void)in_sizes; (void)n_in; (void)out_size; (void)ws_size;
  const float* emb  = (const float*)d_in[0];   // (4, 2048, 1024) fp32
  const float* guid = (const float*)d_in[1];   // (512, 1024) fp32
  float* out = (float*)d_out;                  // (4, 2048, 1024) fp32

  // Workspace layout (bytes):
  //  [0, 2048)                 gscale (512 f32)
  //  [4096, +1MB)              gbf    (512x1024 bf16)
  //  [4096+1MB, +1MB)          gbfT   (1024x512 bf16)
  //  [4096+2MB, +32KB)         mw     (4x2047 f32)
  //  [.., +32MB)               blended(4x2047x1024 f32)
  char* ws = (char*)d_ws;
  float*          gscale  = (float*)(ws);
  unsigned short* gbf     = (unsigned short*)(ws + 4096);
  unsigned short* gbfT    = (unsigned short*)(ws + 4096 + (size_t)G_ * D_ * 2);
  float*          mw      = (float*)(ws + 4096 + 2 * (size_t)G_ * D_ * 2);
  float*          blended = (float*)(ws + 4096 + 2 * (size_t)G_ * D_ * 2 + 32768);

  tisa_prep<<<dim3(G_), dim3(256), 0, stream>>>(guid, gscale, gbf, gbfT);
  tisa_main<<<dim3((T_ + 15) / 16, B_), dim3(256), 0, stream>>>(
      emb, gbf, gbfT, gscale, blended, mw);
  tisa_finalize<<<dim3(S_, B_), dim3(256), 0, stream>>>(blended, mw, out);
}
